// VectorQuantizer_60902636257945
// MI455X (gfx1250) — compile-verified, matrix operand order checked by reading, not run on hardware
//
#include <hip/hip_runtime.h>
#include <hip/hip_bf16.h>

typedef __attribute__((ext_vector_type(16))) _Float16 v16h;
typedef __attribute__((ext_vector_type(8)))  _Float16 v8h;
typedef __attribute__((ext_vector_type(8)))  float    v8f;
typedef __attribute__((ext_vector_type(4)))  int      v4i;

#define K_CODES 1024
#define C_DIM   256
#define HW      1024          // H*W = 32*32
#define N_VECS  32768         // B*H*W
#define M_TILE  128
#define NT      (K_CODES / 16)
#define LDSB_STRIDE 264       // 256 + 8 f16 pad (row = 528B, 16B aligned, 33 x 16B stride)

#if defined(__gfx1250__) && __has_builtin(__builtin_amdgcn_global_load_async_to_lds_b128) && \
    __has_builtin(__builtin_amdgcn_s_wait_asynccnt)
#define VQ_ASYNC 1
typedef __attribute__((address_space(1))) v4i gv4i;   // global int4
typedef __attribute__((address_space(3))) v4i lv4i;   // LDS int4
#else
#define VQ_ASYNC 0
#endif

// ---------------- Kernel 1: split codebook to f16 hi/lo + ||e_k||^2 ----------------
__global__ void vq_prep(const float* __restrict__ emb,
                        _Float16* __restrict__ eh, _Float16* __restrict__ el,
                        float* __restrict__ en2) {
    const int k = blockIdx.x, c = threadIdx.x;
    const float v = emb[k * C_DIM + c];
    const _Float16 hi = (_Float16)v;
    eh[k * C_DIM + c] = hi;
    el[k * C_DIM + c] = (_Float16)(v - (float)hi);
    __shared__ float red[256];
    red[c] = v * v;
    __syncthreads();
    for (int s = 128; s > 0; s >>= 1) {
        if (c < s) red[c] += red[c + s];
        __syncthreads();
    }
    if (c == 0) en2[k] = red[0];
}

// ---------------- Kernel 2: WMMA distance GEMM + running argmin ----------------
__global__ __launch_bounds__(256)
void vq_argmin(const float* __restrict__ z, const _Float16* __restrict__ eh,
               const _Float16* __restrict__ el, const float* __restrict__ en2,
               int* __restrict__ idx_out, float* __restrict__ idxf_out) {
#if VQ_ASYNC
    __shared__ alignas(16) _Float16 sBh[2][16 * LDSB_STRIDE];
    __shared__ alignas(16) _Float16 sBl[2][16 * LDSB_STRIDE];
#else
    __shared__ alignas(16) _Float16 sBh[1][16 * LDSB_STRIDE];
    __shared__ alignas(16) _Float16 sBl[1][16 * LDSB_STRIDE];
#endif

    const int tid  = threadIdx.x;
    const int wave = tid >> 5;
    const int lane = tid & 31;
    const int m    = lane & 15;     // row (A) / column (B) within 16-tile
    const int hsel = lane >> 4;     // which c-half this lane carries

    // This wave's 16 rows of z_flat. n = ((b*32)+h)*32 + w
    const int n_base = blockIdx.x * M_TILE + wave * 16;
    const int n  = n_base + m;
    const int b  = n >> 10;
    const int hh = (n >> 5) & 31;
    const int ww = n & 31;
    const float* zrow = z + (size_t)b * (C_DIM * HW) + hh * 32 + ww;  // + c*HW per channel

    // Load + split A tile into registers, packed in ISA A-fragment lane layout:
    // lane<16: c in {0..7, 16..23}(+32*cc); lane>=16: c in {8..15, 24..31}(+32*cc)
    v16h Ah[8], Al[8];
#pragma unroll
    for (int cc = 0; cc < 8; ++cc) {
#pragma unroll
        for (int e = 0; e < 16; ++e) {
            const int c = cc * 32 + hsel * 8 + (e & 7) + ((e >> 3) << 4);
            const float v = zrow[(size_t)c * HW];
            const _Float16 hi = (_Float16)v;
            Ah[cc][e] = hi;
            Al[cc][e] = (_Float16)(v - (float)hi);
        }
    }

    float minv[8];
    int   mini[8];
#pragma unroll
    for (int r = 0; r < 8; ++r) { minv[r] = 3.4e38f; mini[r] = 0; }

    // ---- per-tile compute: 8 c-chunks x (hh, hl, lh) WMMAs, two acc chains ----
    auto compute_tile = [&](int kt, int buf) {
        v8f acc0 = {}, acc1 = {};
#pragma unroll
        for (int cc = 0; cc < 8; ++cc) {
            const int c0 = cc * 32 + hsel * 8;
            const v8h x0 = *(const v8h*)&sBh[buf][m * LDSB_STRIDE + c0];
            const v8h x1 = *(const v8h*)&sBh[buf][m * LDSB_STRIDE + c0 + 16];
            const v8h y0 = *(const v8h*)&sBl[buf][m * LDSB_STRIDE + c0];
            const v8h y1 = *(const v8h*)&sBl[buf][m * LDSB_STRIDE + c0 + 16];
            v16h Bh, Bl;
#pragma unroll
            for (int i = 0; i < 8; ++i) {
                Bh[i] = x0[i]; Bh[i + 8] = x1[i];
                Bl[i] = y0[i]; Bl[i + 8] = y1[i];
            }
            if (cc & 1) {
                acc1 = __builtin_amdgcn_wmma_f32_16x16x32_f16(false, Ah[cc], false, Bh, (short)0, acc1, false, false);
                acc1 = __builtin_amdgcn_wmma_f32_16x16x32_f16(false, Ah[cc], false, Bl, (short)0, acc1, false, false);
                acc1 = __builtin_amdgcn_wmma_f32_16x16x32_f16(false, Al[cc], false, Bh, (short)0, acc1, false, false);
            } else {
                acc0 = __builtin_amdgcn_wmma_f32_16x16x32_f16(false, Ah[cc], false, Bh, (short)0, acc0, false, false);
                acc0 = __builtin_amdgcn_wmma_f32_16x16x32_f16(false, Ah[cc], false, Bl, (short)0, acc0, false, false);
                acc0 = __builtin_amdgcn_wmma_f32_16x16x32_f16(false, Al[cc], false, Bh, (short)0, acc0, false, false);
            }
        }
        const int   kg = kt * 16 + m;        // this lane's code index (column N)
        const float n2 = en2[kg];
#pragma unroll
        for (int r = 0; r < 8; ++r) {        // acc[r]: row (r + 8*hsel), col kg
            const float s = n2 - 2.0f * (acc0[r] + acc1[r]);
            const bool better = s < minv[r];
            mini[r] = better ? kg : mini[r];
            minv[r] = better ? s : minv[r];
        }
    };

#if VQ_ASYNC
    // ---- async Global->LDS DMA pipeline, double-buffered B tile ----
    // 256 thr cover 2x 8KB tile: thread t handles 16B at rows r0 (bytes [0,4K)) and r0+8.
    const int to  = tid << 4;        // byte offset within 4KB half
    const int ar0 = to >> 9;         // row 0..7 (global row stride 512B)
    const int acb = to & 511;        // column bytes
    const char* geh = (const char*)eh;
    const char* gel = (const char*)el;
    auto issue = [&](int kt, int buf) {
        const size_t g0 = (size_t)(kt * 16 + ar0) * 512 + acb;
        const size_t g1 = (size_t)(kt * 16 + ar0 + 8) * 512 + acb;
        char* d0h = (char*)&sBh[buf][0] + ar0 * (LDSB_STRIDE * 2) + acb;
        char* d1h = (char*)&sBh[buf][0] + (ar0 + 8) * (LDSB_STRIDE * 2) + acb;
        char* d0l = (char*)&sBl[buf][0] + ar0 * (LDSB_STRIDE * 2) + acb;
        char* d1l = (char*)&sBl[buf][0] + (ar0 + 8) * (LDSB_STRIDE * 2) + acb;
        __builtin_amdgcn_global_load_async_to_lds_b128((gv4i*)(geh + g0), (lv4i*)d0h, 0, 0);
        __builtin_amdgcn_global_load_async_to_lds_b128((gv4i*)(geh + g1), (lv4i*)d1h, 0, 0);
        __builtin_amdgcn_global_load_async_to_lds_b128((gv4i*)(gel + g0), (lv4i*)d0l, 0, 0);
        __builtin_amdgcn_global_load_async_to_lds_b128((gv4i*)(gel + g1), (lv4i*)d1l, 0, 0);
    };
    issue(0, 0);
    for (int kt = 0; kt < NT; ++kt) {
        const int cur = kt & 1;
        if (kt + 1 < NT) {
            issue(kt + 1, cur ^ 1);                  // DMA next tile while computing this one
            __builtin_amdgcn_s_wait_asynccnt(4);     // in-order: <=4 outstanding => tile kt landed
        } else {
            __builtin_amdgcn_s_wait_asynccnt(0);
        }
        __syncthreads();
        compute_tile(kt, cur);
        __syncthreads();                             // done reading buf[cur] before it is DMA'd again
    }
#else
    // ---- fallback: register-staged software pipeline (issue kt+1 before computing kt) ----
    const int bl_k = tid >> 4;
    const int bl_c = (tid & 15) << 4;
    v8h r0, r1, r2, r3;
    auto stage_load = [&](int kt) {
        const size_t ro = (size_t)(kt * 16 + bl_k) * C_DIM + bl_c;
        r0 = *(const v8h*)(eh + ro);
        r1 = *(const v8h*)(eh + ro + 8);
        r2 = *(const v8h*)(el + ro);
        r3 = *(const v8h*)(el + ro + 8);
    };
    stage_load(0);
    for (int kt = 0; kt < NT; ++kt) {
        __syncthreads();                             // everyone done reading previous tile
        *(v8h*)&sBh[0][bl_k * LDSB_STRIDE + bl_c]     = r0;
        *(v8h*)&sBh[0][bl_k * LDSB_STRIDE + bl_c + 8] = r1;
        *(v8h*)&sBl[0][bl_k * LDSB_STRIDE + bl_c]     = r2;
        *(v8h*)&sBl[0][bl_k * LDSB_STRIDE + bl_c + 8] = r3;
        if (kt + 1 < NT) {
            stage_load(kt + 1);                      // latency hidden behind compute_tile(kt)
            __builtin_prefetch(eh + (size_t)((kt + 2) * 16 + bl_k) * C_DIM + bl_c, 0, 1);
            __builtin_prefetch(el + (size_t)((kt + 2) * 16 + bl_k) * C_DIM + bl_c, 0, 1);
        }
        __syncthreads();
        compute_tile(kt, 0);
    }
#endif

    // reduce (min, argmin) across the 16 lanes sharing each row; xor<=8 stays in-half
#pragma unroll
    for (int r = 0; r < 8; ++r) {
#pragma unroll
        for (int off = 1; off < 16; off <<= 1) {
            const float ov = __shfl_xor(minv[r], off, 32);
            const int   oi = __shfl_xor(mini[r], off, 32);
            const bool take = (ov < minv[r]) || (ov == minv[r] && oi < mini[r]);
            minv[r] = take ? ov : minv[r];
            mini[r] = take ? oi : mini[r];
        }
    }
    if (m == 0) {                            // lanes 0 (rows 0..7) and 16 (rows 8..15)
        const int base = n_base + hsel * 8;
#pragma unroll
        for (int r = 0; r < 8; ++r) {
            idx_out[base + r]  = mini[r];
            idxf_out[base + r] = (float)mini[r];
        }
    }
    (void)wave;
}

// ---------------- Kernel 3: gather zq (BCHW), deterministic partial MSE ----------------
__global__ void vq_apply(const float* __restrict__ z, const float* __restrict__ emb,
                         const int* __restrict__ idx, float* __restrict__ zq,
                         float* __restrict__ part) {
    const int blk = blockIdx.x;        // b*32 + h
    const int b = blk >> 5, h = blk & 31;
    const int w  = threadIdx.x & 31;
    const int cg = threadIdx.x >> 5;   // 0..7
    const int n  = (b * 32 + h) * 32 + w;
    const int id = idx[n];
    const float* er = emb + (size_t)id * C_DIM;
    const size_t zbase = (size_t)b * (C_DIM * HW) + h * 32 + w;

    float acc = 0.f;
#pragma unroll 4
    for (int j = 0; j < 32; ++j) {
        const int c = cg * 32 + j;
        const float ev = er[c];
        const size_t o = zbase + (size_t)c * HW;
        const float d = ev - z[o];
        acc += d * d;
        zq[o] = ev;                     // straight-through output == e[idx]
    }
    __shared__ float red[256];
    red[threadIdx.x] = acc;
    __syncthreads();
    for (int s = 128; s > 0; s >>= 1) {
        if (threadIdx.x < s) red[threadIdx.x] += red[threadIdx.x + s];
        __syncthreads();
    }
    if (threadIdx.x == 0) part[blk] = red[0];
}

// ---------------- Kernel 4: fixed-order final reduction -> 3 scalars ----------------
__global__ void vq_finalize(const float* __restrict__ part, float* __restrict__ out3) {
    __shared__ float red[256];
    const int t = threadIdx.x;
    red[t] = part[t] + part[t + 256] + part[t + 512] + part[t + 768];
    __syncthreads();
    for (int s = 128; s > 0; s >>= 1) {
        if (t < s) red[t] += red[t + s];
        __syncthreads();
    }
    if (t == 0) {
        const float mse = red[0] / 8388608.0f;   // N*C = 2^23
        out3[0] = 1.25f * mse;                   // loss
        out3[1] = 0.25f * mse;                   // commitment_loss
        out3[2] = mse;                           // codebook_loss
    }
}

extern "C" void kernel_launch(void* const* d_in, const int* in_sizes, int n_in,
                              void* d_out, int out_size, void* d_ws, size_t ws_size,
                              hipStream_t stream) {
    (void)in_sizes; (void)n_in; (void)out_size; (void)ws_size;
    const float* z   = (const float*)d_in[0];
    const float* emb = (const float*)d_in[1];
    float* out = (float*)d_out;

    char* ws = (char*)d_ws;
    _Float16* eh  = (_Float16*)(ws);                                   // 512 KB
    _Float16* el  = (_Float16*)(ws + 512 * 1024);                      // 512 KB
    float*    en2 = (float*)   (ws + 1024 * 1024);                     // 4 KB
    int*      idx = (int*)     (ws + 1024 * 1024 + 4096);              // 128 KB
    float*    prt = (float*)   (ws + 1024 * 1024 + 4096 + 128 * 1024); // 4 KB

    float* zq    = out;                 // [32,256,32,32]
    float* loss3 = out + 8388608;       // loss, commitment, codebook
    float* idxf  = out + 8388611;       // indices as float, [32,32,32]

    vq_prep    <<<K_CODES,       256, 0, stream>>>(emb, eh, el, en2);
    vq_argmin  <<<N_VECS/M_TILE, 256, 0, stream>>>(z, eh, el, en2, idx, idxf);
    vq_apply   <<<HW,            256, 0, stream>>>(z, emb, idx, zq, prt);
    vq_finalize<<<1,             256, 0, stream>>>(prt, loss3);
}
